// AttnSpatioTemporalDecoder_26027501814591
// MI455X (gfx1250) — matrix operand order repaired and checked by reading, not compiled, not run on hardware
//
#include <hip/hip_runtime.h>
#include <hip/hip_bf16.h>

// ---------------------------------------------------------------------------
// AttnSpatioTemporalDecoder for MI455X (gfx1250, wave32, WMMA).
// B=8, N=50, T=15, F_IN=6, K=4, H=128, A=64, D=134, E=2450, M=B*E=19600.
// Heavy GEMMs: v_wmma_f32_16x16x32_bf16, 5x2 register-blocked (10 C tiles
// per wave) to raise L0 arithmetic intensity ~3x vs 1-tile-per-wave.
// ---------------------------------------------------------------------------

#define BB 8
#define NN 50
#define TT 15
#define FIN 6
#define KK 4
#define HH 128
#define AA 64
#define DD 134          // FIN + HH
#define EE 2450         // N*(N-1)
#define ME (BB*EE)      // 19600 = 80 * 245  (exact 5x16-row blocks)
#define MN (BB*NN)      // 400
#define K1PAD 288       // 2*D = 268 padded to mult of 32
#define G3H 384         // 3*H
#define G3D 402         // 3*D

typedef __attribute__((ext_vector_type(16))) __bf16 v16bf;
typedef __attribute__((ext_vector_type(8)))  float  v8f;

union Frag { v16bf v; int4 q[2]; };

__device__ __forceinline__ float sigf(float x) { return 1.0f / (1.0f + expf(-x)); }

// ---------------------------------------------------------------------------
// Weight prep: fp32 [seg][Ksrc][Hseg] -> bf16 transposed/padded [Ncols][Kpad]
// ---------------------------------------------------------------------------
__global__ void wprep_kernel(const float* __restrict__ src, __bf16* __restrict__ dst,
                             int Ncols, int Kpad, int Ksrc, int Hseg, long segPitch) {
    int idx = blockIdx.x * blockDim.x + threadIdx.x;
    if (idx >= Ncols * Kpad) return;
    int c = idx / Kpad, kk = idx % Kpad;
    float v = 0.0f;
    if (kk < Ksrc) v = src[(long)(c / Hseg) * segPitch + (long)kk * Hseg + (c % Hseg)];
    dst[idx] = (__bf16)v;
}

// ---------------------------------------------------------------------------
// bf16 WMMA GEMM: C[M x Ncols] = act(A @ W + bias), 5 (tm) x 2 (tn) tiles/wave.
//   A:  bf16 row-major [M][lda], base += (colBlock32/aseg)*asegPitch (k banks)
//   Wt: bf16 [Ncols][ldw] (W transposed: a column of W contiguous in K)
//   out offset = (col/oseg)*osegPitch + row*ldc + col%oseg  (fp32 and/or bf16)
// Requires M % 80 == 0, Ncols % 32 == 0, Kpad % 32 == 0, aseg/oseg mult of 32.
// Fragment layouts per CDNA5 ISA 7.12.2 (16-bit A 16x32, B 32x16, C 16x16 f32).
// ---------------------------------------------------------------------------
__global__ __launch_bounds__(256) void wmma_gemm(
    const __bf16* __restrict__ A, int lda, long asegPitch, int aseg,
    const __bf16* __restrict__ Wt, int ldw,
    const float* __restrict__ bias,
    float* __restrict__ outF, __bf16* __restrict__ outB,
    int ldc, long osegPitch, int oseg,
    int M, int Kpad, int Ncols, int act)
{
    int wave = blockIdx.x * (blockDim.x >> 5) + (threadIdx.x >> 5);
    int lane = threadIdx.x & 31;
    int nb = Ncols >> 5;          // 32-column blocks
    int mb = M / 80;              // 80-row blocks
    if (wave >= mb * nb) return;  // wave-uniform
    int bm = wave / nb, bn = wave % nb;
    bool hi = lane >= 16;
    int l15 = lane & 15;
    int ha = hi ? 8 : 0;          // A frag: lanes 16-31 hold K+8..15 / K+24..31
    int hb = hi ? 8 : 0;          // B frag: lanes 16-31 hold K+8..15 / K+24..31

    const __bf16* Abase = A + (long)((bn * 32) / aseg) * asegPitch;
    const __bf16* Arow[5];
#pragma unroll
    for (int u = 0; u < 5; ++u)
        Arow[u] = Abase + (long)(bm * 80 + u * 16 + l15) * lda + ha;
    const __bf16* Wrow[2];
#pragma unroll
    for (int v = 0; v < 2; ++v)
        Wrow[v] = Wt + (long)(bn * 32 + v * 16 + l15) * ldw + hb;

    v8f acc[5][2] = {};
    for (int k0 = 0; k0 < Kpad; k0 += 32) {
        Frag a[5], b[2];
#pragma unroll
        for (int u = 0; u < 5; ++u) {
            a[u].q[0] = *(const int4*)(Arow[u] + k0);       // K 0..7   (+ha)
            a[u].q[1] = *(const int4*)(Arow[u] + k0 + 16);  // K 16..23 (+ha)
        }
#pragma unroll
        for (int v = 0; v < 2; ++v) {
            b[v].q[0] = *(const int4*)(Wrow[v] + k0);       // K 0..7   (+hb)
            b[v].q[1] = *(const int4*)(Wrow[v] + k0 + 16);  // K 16..23 (+hb)
        }
#pragma unroll
        for (int u = 0; u < 5; ++u)
#pragma unroll
            for (int v = 0; v < 2; ++v)
                acc[u][v] = __builtin_amdgcn_wmma_f32_16x16x32_bf16(
                    false, a[u].v, false, b[v].v, (short)0, acc[u][v], false, false);
    }

#pragma unroll
    for (int v = 0; v < 2; ++v) {
        int col = bn * 32 + v * 16 + l15;
        float bc = bias ? bias[col] : 0.0f;
        int ks = col / oseg, cc = col % oseg;
#pragma unroll
        for (int u = 0; u < 5; ++u) {
            int rbase = bm * 80 + u * 16 + (hi ? 8 : 0);   // C: hi lanes = rows +8
#pragma unroll
            for (int r = 0; r < 8; ++r) {
                float x = acc[u][v][r] + bc;
                if (act == 1)      x = tanhf(x);
                else if (act == 2) x = fmaxf(x, 0.0f);
                long off = (long)ks * osegPitch + (long)(rbase + r) * ldc + cc;
                if (outF) outF[off] = x;
                if (outB) outB[off] = (__bf16)x;
            }
        }
    }
}

// ---------------------------------------------------------------------------
// x[b][n][0:128] = ins @ emb_w + emb_b ; x[b][n][128:134] = ins   (bf16)
// ---------------------------------------------------------------------------
__global__ void embed_kernel(const float* __restrict__ data,
                             const float* __restrict__ ew, const float* __restrict__ eb,
                             __bf16* __restrict__ x, int step) {
    int idx = blockIdx.x * blockDim.x + threadIdx.x;
    if (idx >= MN * DD) return;
    int bn = idx / DD, c = idx % DD;
    const float* ins = data + ((long)bn * TT + step) * FIN;
    float v;
    if (c < HH) {
        v = eb[c];
        for (int f = 0; f < FIN; ++f) v += ins[f] * ew[f * HH + c];
    } else {
        v = ins[c - HH];
    }
    x[idx] = (__bf16)v;
}

// pre_msg[b][e][0:134]=x[send], [134:268]=x[recv], [268:288]=0   (bf16)
__global__ void gather_kernel(const __bf16* __restrict__ x, __bf16* __restrict__ pm) {
    int idx = blockIdx.x * blockDim.x + threadIdx.x;
    if (idx >= ME * K1PAD) return;
    int m = idx / K1PAD, c = idx % K1PAD;
    int b = m / EE, e = m % EE;
    int i = e / (NN - 1), jj = e % (NN - 1);
    int j = jj + (jj >= i ? 1 : 0);          // send=i, recv=j
    __bf16 v;
    if (c < DD)            v = x[(b * NN + i) * DD + c];
    else if (c < 2 * DD)   v = x[(b * NN + j) * DD + (c - DD)];
    else                   v = (__bf16)0.0f;
    pm[idx] = v;
}

// all_msgs = (sum_k m2[:,k,:] * rel_type[:,:,k]) / K     (m2 already tanh'ed)
__global__ void msum_kernel(const float* __restrict__ m2, const float* __restrict__ rel,
                            float* __restrict__ outF, __bf16* __restrict__ outB) {
    int idx = blockIdx.x * blockDim.x + threadIdx.x;
    if (idx >= ME * HH) return;
    int m = idx / HH, h = idx % HH;
    float s = 0.0f;
    for (int k = 0; k < KK; ++k)
        s += m2[(long)m * (KK * HH) + k * HH + h] * rel[m * KK + k];
    s *= 0.25f;
    outF[idx] = s;
    outB[idx] = (__bf16)s;
}

// Edge GRU combine: h' = (1-z)*n + z*h  (in-place on hF), also bf16 copy
__global__ void gru_edge_kernel(const float* __restrict__ gx, const float* __restrict__ gh,
                                float* __restrict__ hF, __bf16* __restrict__ hB) {
    int idx = blockIdx.x * blockDim.x + threadIdx.x;
    if (idx >= ME * HH) return;
    int m = idx / HH, h = idx % HH;
    long o = (long)m * G3H + h;
    float r = sigf(gx[o] + gh[o]);
    float z = sigf(gx[o + HH] + gh[o + HH]);
    float n = tanhf(gx[o + 2 * HH] + r * gh[o + 2 * HH]);
    float hv = (1.0f - z) * n + z * hF[idx];
    hF[idx] = hv;
    hB[idx] = (__bf16)hv;
}

// agg[b][n][h] = sum_{i != n} msgs[b][e(i,n)][h] / F_IN
__global__ void agg_kernel(const float* __restrict__ msgs, float* __restrict__ agg) {
    int idx = blockIdx.x * blockDim.x + threadIdx.x;
    if (idx >= MN * HH) return;
    int bn = idx / HH, h = idx % HH;
    int b = bn / NN, n = bn % NN;
    float s = 0.0f;
    for (int i = 0; i < NN; ++i) {
        if (i == n) continue;
        int jj = (n < i) ? n : n - 1;
        int e = i * (NN - 1) + jj;
        s += msgs[((long)(b * EE + e)) * HH + h];
    }
    agg[idx] = s * (1.0f / (float)FIN);
}

// cat = [ins ; agg]; on step 0 also seed h_node and h_attn[0]
__global__ void cat_kernel(const float* __restrict__ data, const float* __restrict__ agg,
                           float* __restrict__ cat, float* __restrict__ hn,
                           float* __restrict__ hattn, int step) {
    int idx = blockIdx.x * blockDim.x + threadIdx.x;
    if (idx >= MN * DD) return;
    int m = idx / DD, c = idx % DD;
    float v = (c < FIN) ? data[((long)m * TT + step) * FIN + c]
                        : agg[m * HH + (c - FIN)];
    cat[idx] = v;
    if (step == 0) { hn[idx] = v; hattn[idx] = v; }
}

// Node GRU gate GEMM (tiny): out[m][c] = bias[c] + in[m] . w[:,c]
__global__ void node_gates_kernel(const float* __restrict__ in, const float* __restrict__ w,
                                  const float* __restrict__ bias, float* __restrict__ out) {
    int idx = blockIdx.x * blockDim.x + threadIdx.x;
    if (idx >= MN * G3D) return;
    int m = idx / G3D, c = idx % G3D;
    float a = bias[c];
    const float* r = in + (long)m * DD;
    for (int d = 0; d < DD; ++d) a += r[d] * w[d * G3D + c];
    out[idx] = a;
}

__global__ void node_combine_kernel(const float* __restrict__ gx, const float* __restrict__ gh,
                                    float* __restrict__ hn) {
    int idx = blockIdx.x * blockDim.x + threadIdx.x;
    if (idx >= MN * DD) return;
    int m = idx / DD, c = idx % DD;
    long o = (long)m * G3D + c;
    float r = sigf(gx[o] + gh[o]);
    float z = sigf(gx[o + DD] + gh[o + DD]);
    float n = tanhf(gx[o + 2 * DD] + r * gh[o + 2 * DD]);
    hn[idx] = (1.0f - z) * n + z * hn[idx];
}

// Temporal attention; block per (b,n). S = step+1 keys (hist + current).
__global__ __launch_bounds__(192) void attn_kernel(
    const float* __restrict__ hn, float* __restrict__ hattn,
    const float* __restrict__ qw, const float* __restrict__ qb,
    const float* __restrict__ kw, const float* __restrict__ kb,
    const float* __restrict__ vw, const float* __restrict__ vb,
    const float* __restrict__ aow, const float* __restrict__ aob,
    float* __restrict__ cur, int step)
{
    int m = blockIdx.x, t = threadIdx.x;
    int S = step + 1;
    __shared__ float h[DD];
    __shared__ float q[AA];
    __shared__ float kk[(TT - 1) * AA];
    __shared__ float vv[(TT - 1) * AA];
    __shared__ float sc[TT];
    __shared__ float av[AA];
    for (int c = t; c < DD; c += 192) h[c] = hn[(long)m * DD + c];
    __syncthreads();
    if (t < AA) {
        float a = qb[t];
        for (int d = 0; d < DD; ++d) a += h[d] * qw[d * AA + t];
        q[t] = fmaxf(a, 0.0f);
    }
    for (int idx = t; idx < S * AA; idx += 192) {
        int s = idx >> 6, a = idx & (AA - 1);
        const float* src = (s < step) ? (hattn + ((long)s * MN + m) * DD) : nullptr;
        float ka = kb[a], va = vb[a];
        for (int d = 0; d < DD; ++d) {
            float x = src ? src[d] : h[d];
            ka += x * kw[d * AA + a];
            va += x * vw[d * AA + a];
        }
        kk[idx] = fmaxf(ka, 0.0f);
        vv[idx] = fmaxf(va, 0.0f);
    }
    __syncthreads();
    if (t < S) {
        float d = 0.0f;
        for (int a = 0; a < AA; ++a) d += q[a] * kk[t * AA + a];
        sc[t] = d * 0.125f;                  // 1/sqrt(A)
    }
    __syncthreads();
    if (t < AA) {
        float mx = -1e30f;
        for (int s = 0; s < S; ++s) mx = fmaxf(mx, sc[s]);
        float den = 0.0f, ac = 0.0f;
        for (int s = 0; s < S; ++s) {
            float w = expf(sc[s] - mx);
            den += w; ac += w * vv[s * AA + t];
        }
        av[t] = ac / den;
    }
    __syncthreads();
    for (int c = t; c < DD; c += 192) {
        float o = aob[c];
        for (int a = 0; a < AA; ++a) o += av[a] * aow[a * DD + c];
        o = fmaxf(o, 0.0f);
        cur[(long)m * DD + c] = o;
        hattn[((long)step * MN + m) * DD + c] = o;   // append
    }
}

// Output MLP + residual; block per (b,n).
__global__ __launch_bounds__(128) void out_mlp_kernel(
    const float* __restrict__ z0buf, const float* __restrict__ data,
    const float* __restrict__ w1, const float* __restrict__ b1,
    const float* __restrict__ w2, const float* __restrict__ b2,
    const float* __restrict__ w3, const float* __restrict__ b3,
    float* __restrict__ out, int step)
{
    int m = blockIdx.x, t = threadIdx.x;
    __shared__ float z0[DD], z1[HH], z2[HH];
    for (int c = t; c < DD; c += 128) z0[c] = z0buf[(long)m * DD + c];
    __syncthreads();
    float acc = b1[t];
    for (int d = 0; d < DD; ++d) acc += z0[d] * w1[d * HH + t];
    z1[t] = fmaxf(acc, 0.0f);
    __syncthreads();
    acc = b2[t];
    for (int d = 0; d < HH; ++d) acc += z1[d] * w2[d * HH + t];
    z2[t] = fmaxf(acc, 0.0f);
    __syncthreads();
    if (t < FIN) {
        float o = b3[t];
        for (int d = 0; d < HH; ++d) o += z2[d] * w3[d * FIN + t];
        float ins = data[((long)m * TT + step) * FIN + t];
        out[((long)m * (TT - 1) + step) * FIN + t] = ins + o;  // [B][N][T-1][F]
    }
}

// ---------------------------------------------------------------------------
static inline int cdiv(int a, int b) { return (a + b - 1) / b; }

extern "C" void kernel_launch(void* const* d_in, const int* in_sizes, int n_in,
                              void* d_out, int out_size, void* d_ws, size_t ws_size,
                              hipStream_t stream) {
    (void)in_sizes; (void)n_in; (void)out_size; (void)ws_size;
    const float* data     = (const float*)d_in[0];   // [B][N][T][F]
    const float* rel_type = (const float*)d_in[1];   // [B][E][K]
    // d_in[2]=rel_rec, d_in[3]=rel_send: one-hot of analytic indices -> unused
    const float* emb_w = (const float*)d_in[4];
    const float* emb_b = (const float*)d_in[5];
    const float* msg_w1 = (const float*)d_in[6];     // [K][268][128]
    const float* msg_b1 = (const float*)d_in[7];     // [K][128] -> flat 512
    const float* msg_w2 = (const float*)d_in[8];     // [K][128][128]
    const float* msg_b2 = (const float*)d_in[9];
    const float* gru_e_wx = (const float*)d_in[10];  // [128][384]
    const float* gru_e_bx = (const float*)d_in[11];
    const float* gru_e_wh = (const float*)d_in[12];
    const float* gru_e_bh = (const float*)d_in[13];
    const float* gru_n_wx = (const float*)d_in[14];  // [134][402]
    const float* gru_n_bx = (const float*)d_in[15];
    const float* gru_n_wh = (const float*)d_in[16];
    const float* gru_n_bh = (const float*)d_in[17];
    const float* out1_w = (const float*)d_in[18];
    const float* out1_b = (const float*)d_in[19];
    const float* out2_w = (const float*)d_in[20];
    const float* out2_b = (const float*)d_in[21];
    const float* out3_w = (const float*)d_in[22];
    const float* out3_b = (const float*)d_in[23];
    const float* q_w = (const float*)d_in[24];
    const float* q_b = (const float*)d_in[25];
    const float* k_w = (const float*)d_in[26];
    const float* k_b = (const float*)d_in[27];
    const float* v_w = (const float*)d_in[28];
    const float* v_b = (const float*)d_in[29];
    const float* ao_w = (const float*)d_in[30];
    const float* ao_b = (const float*)d_in[31];
    float* out = (float*)d_out;

    // --- workspace bump allocator -------------------------------------------
    char* ws = (char*)d_ws;
    size_t off = 0;
    auto alloc = [&](size_t bytes) -> void* {
        off = (off + 255) & ~(size_t)255;
        void* p = ws + off;
        off += bytes;
        return p;
    };
    __bf16* w1t  = (__bf16*)alloc((size_t)(KK * HH) * K1PAD * 2);   // [512][288]
    __bf16* w2t  = (__bf16*)alloc((size_t)(KK * HH) * HH * 2);      // [512][128]
    __bf16* wxet = (__bf16*)alloc((size_t)G3H * HH * 2);            // [384][128]
    __bf16* whet = (__bf16*)alloc((size_t)G3H * HH * 2);
    __bf16* xbf  = (__bf16*)alloc((size_t)MN * DD * 2);
    __bf16* pm   = (__bf16*)alloc((size_t)ME * K1PAD * 2);
    __bf16* m1b  = (__bf16*)alloc((size_t)KK * ME * HH * 2);        // [k][M][128]
    float*  m2f  = (float*)alloc((size_t)ME * KK * HH * 4);         // [M][512]
    float*  allF = (float*)alloc((size_t)ME * HH * 4);
    __bf16* allB = (__bf16*)alloc((size_t)ME * HH * 2);
    float*  gx   = (float*)alloc((size_t)ME * G3H * 4);
    float*  gh   = (float*)alloc((size_t)ME * G3H * 4);
    float*  heF  = (float*)alloc((size_t)ME * HH * 4);              // h_edge / msgs
    __bf16* heB  = (__bf16*)alloc((size_t)ME * HH * 2);
    float*  agg  = (float*)alloc((size_t)MN * HH * 4);
    float*  catF = (float*)alloc((size_t)MN * DD * 4);
    float*  hnF  = (float*)alloc((size_t)MN * DD * 4);
    float*  gxn  = (float*)alloc((size_t)MN * G3D * 4);
    float*  ghn  = (float*)alloc((size_t)MN * G3D * 4);
    float*  hattn= (float*)alloc((size_t)(TT - 1) * MN * DD * 4);
    float*  curF = (float*)alloc((size_t)MN * DD * 4);

    // --- one-time weight prep (bf16, transposed, K-padded) ------------------
    wprep_kernel<<<cdiv(KK * HH * K1PAD, 256), 256, 0, stream>>>(
        msg_w1, w1t, KK * HH, K1PAD, 2 * DD, HH, (long)(2 * DD) * HH);
    wprep_kernel<<<cdiv(KK * HH * HH, 256), 256, 0, stream>>>(
        msg_w2, w2t, KK * HH, HH, HH, HH, (long)HH * HH);
    wprep_kernel<<<cdiv(G3H * HH, 256), 256, 0, stream>>>(
        gru_e_wx, wxet, G3H, HH, HH, G3H, 0);
    wprep_kernel<<<cdiv(G3H * HH, 256), 256, 0, stream>>>(
        gru_e_wh, whet, G3H, HH, HH, G3H, 0);

    const int wavesMsg = (ME / 80) * ((KK * HH) / 32);   // 245*16 = 3920
    const int wavesGru = (ME / 80) * (G3H / 32);         // 245*12 = 2940

    for (int step = 0; step < TT - 1; ++step) {
        // 1) node embedding -> x (bf16)
        embed_kernel<<<cdiv(MN * DD, 256), 256, 0, stream>>>(data, emb_w, emb_b, xbf, step);
        // 2) gather send/recv -> pre_msg (bf16, K-padded to 288)
        gather_kernel<<<cdiv(ME * K1PAD, 256), 256, 0, stream>>>(xbf, pm);
        // 3) msg MLP1: tanh(pre_msg @ w1[k] + b1[k]) -> m1b [k][M][128]  (WMMA)
        wmma_gemm<<<cdiv(wavesMsg, 8), 256, 0, stream>>>(
            pm, K1PAD, 0, KK * HH, w1t, K1PAD, msg_b1,
            nullptr, m1b, HH, (long)ME * HH, HH, ME, K1PAD, KK * HH, 1);
        // 4) msg MLP2: tanh(m1[k] @ w2[k] + b2[k]) -> m2f [M][512]       (WMMA)
        wmma_gemm<<<cdiv(wavesMsg, 8), 256, 0, stream>>>(
            m1b, HH, (long)ME * HH, HH, w2t, HH, msg_b2,
            m2f, nullptr, KK * HH, 0, KK * HH, ME, HH, KK * HH, 1);
        // 5) rel_type-weighted sum over k
        if (step == 0) {
            msum_kernel<<<cdiv(ME * HH, 256), 256, 0, stream>>>(m2f, rel_type, heF, heB);
        } else {
            msum_kernel<<<cdiv(ME * HH, 256), 256, 0, stream>>>(m2f, rel_type, allF, allB);
            // 6) edge GRU gates (WMMA) + combine
            wmma_gemm<<<cdiv(wavesGru, 8), 256, 0, stream>>>(
                allB, HH, 0, G3H, wxet, HH, gru_e_bx,
                gx, nullptr, G3H, 0, G3H, ME, HH, G3H, 0);
            wmma_gemm<<<cdiv(wavesGru, 8), 256, 0, stream>>>(
                heB, HH, 0, G3H, whet, HH, gru_e_bh,
                gh, nullptr, G3H, 0, G3H, ME, HH, G3H, 0);
            gru_edge_kernel<<<cdiv(ME * HH, 256), 256, 0, stream>>>(gx, gh, heF, heB);
        }
        // 7) scatter-aggregate messages to receivers
        agg_kernel<<<cdiv(MN * HH, 256), 256, 0, stream>>>(heF, agg);
        // 8) cat = [ins ; agg]; seed h_node/h_attn on step 0
        cat_kernel<<<cdiv(MN * DD, 256), 256, 0, stream>>>(data, agg, catF, hnF, hattn, step);
        if (step > 0) {
            // 9) node GRU (tiny) + temporal attention
            node_gates_kernel<<<cdiv(MN * G3D, 256), 256, 0, stream>>>(catF, gru_n_wx, gru_n_bx, gxn);
            node_gates_kernel<<<cdiv(MN * G3D, 256), 256, 0, stream>>>(hnF, gru_n_wh, gru_n_bh, ghn);
            node_combine_kernel<<<cdiv(MN * DD, 256), 256, 0, stream>>>(gxn, ghn, hnF);
            attn_kernel<<<MN, 192, 0, stream>>>(hnF, hattn, q_w, q_b, k_w, k_b,
                                                v_w, v_b, ao_w, ao_b, curF, step);
        }
        // 10) output MLP + residual -> d_out[b][n][step][f]
        out_mlp_kernel<<<MN, 128, 0, stream>>>(
            (step == 0) ? catF : curF, data,
            out1_w, out1_b, out2_w, out2_b, out3_w, out3_b, out, step);
    }
}
